// LSTMClassifier_12996571038386
// MI455X (gfx1250) — matrix operand order repaired and checked by reading, not compiled, not run on hardware
//
#include <hip/hip_runtime.h>
#include <cstdint>

// ---------------------------------------------------------------------------
// Dimensions
// ---------------------------------------------------------------------------
#define BATCH 64
#define TSEQ  2000
#define FIN   98
#define HID   128
#define G4H   512          // 4*H
#define EMB   128
#define NCLS  2
#define MROWS (TSEQ * BATCH)   // 128000 rows of (t,b)

typedef __attribute__((ext_vector_type(16))) _Float16 v16h;
typedef __attribute__((ext_vector_type(8)))  _Float16 v8h;
typedef __attribute__((ext_vector_type(8)))  float    v8f;

// Hardware tanh (TRANS32 on CDNA5). Fallback: inline asm v_tanh_f32 with two
// v_nops to satisfy the TRANS-op result hazard (ISA 7.4: one independent op
// or V_NOP after a trans op before its output is consumed).
__device__ __forceinline__ float tanh_hw(float x) {
#if __has_builtin(__builtin_amdgcn_tanhf)
    return __builtin_amdgcn_tanhf(x);
#elif __has_builtin(__builtin_amdgcn_tanh_f32)
    return __builtin_amdgcn_tanh_f32(x);
#else
    float r;
    asm volatile("v_tanh_f32 %0, %1\n\tv_nop\n\tv_nop" : "=v"(r) : "v"(x));
    return r;
#endif
}

// sigmoid via hardware tanh: sig(x) = 0.5 + 0.5*tanh(x/2)
__device__ __forceinline__ float sigmoid_hw(float x) {
    return fmaf(0.5f, tanh_hw(0.5f * x), 0.5f);
}

// Build a 16-element f16 WMMA fragment from two contiguous 8-halves chunks.
__device__ __forceinline__ v16h make_frag(const _Float16* lo_p, const _Float16* hi_p) {
    v8h lo = *(const v8h*)lo_p;
    v8h hi = *(const v8h*)hi_p;
    v16h r;
#pragma unroll
    for (int i = 0; i < 8; ++i) { r[i] = lo[i]; r[i + 8] = hi[i]; }
    return r;
}

// ---------------------------------------------------------------------------
// Conversion / prep kernels
// ---------------------------------------------------------------------------

// x [B][1][T][F] f32  ->  A0 [(t*64+b)][128] f16, zero-padded K 98->128
__global__ void convert_x_kernel(const float* __restrict__ x, _Float16* __restrict__ A0) {
    const size_t n = (size_t)MROWS * 128;
    for (size_t i = (size_t)blockIdx.x * blockDim.x + threadIdx.x; i < n;
         i += (size_t)gridDim.x * blockDim.x) {
        int    k   = (int)(i & 127);
        size_t row = i >> 7;
        int    b   = (int)(row & 63);
        int    t   = (int)(row >> 6);
        A0[i] = (k < FIN) ? (_Float16)x[((size_t)b * TSEQ + t) * FIN + k] : (_Float16)0.f;
    }
}

// src [512][Ks] f32 -> dst [512][KD] f16, zero-padded (KD power of two)
template <int KD>
__global__ void convert_w_kernel(const float* __restrict__ src, _Float16* __restrict__ dst,
                                 int Ks) {
    const size_t n = (size_t)G4H * KD;
    for (size_t i = (size_t)blockIdx.x * blockDim.x + threadIdx.x; i < n;
         i += (size_t)gridDim.x * blockDim.x) {
        int k = (int)(i & (KD - 1));
        int r = (int)(i / KD);             // KD constant -> shift
        dst[i] = (k < Ks) ? (_Float16)src[(size_t)r * Ks + k] : (_Float16)0.f;
    }
}

__global__ void bias_kernel(const float* __restrict__ bi, const float* __restrict__ bh,
                            float* __restrict__ out) {
    int i = blockIdx.x * blockDim.x + threadIdx.x;
    if (i < G4H) out[i] = bi[i] + bh[i];
}

// ---------------------------------------------------------------------------
// Input-projection GEMM:  xg[t][j][b] = sum_k A[t*64+b][k] * W[j][k] + bias[j]
// A: [M][K] f16 row-major, W: [512][K] f16, out f32 in [T][512][64] layout so
// the per-lane 8 consecutive batch rows of a WMMA C tile are contiguous.
// One WG = one time step (64 rows) x 512 cols, 32 waves, 4 tiles per wave.
// A tile staged into LDS via async global->LDS copies (ASYNCcnt).
// ---------------------------------------------------------------------------
template <int K>
__global__ __launch_bounds__(1024) void xproj_gemm_kernel(
    const _Float16* __restrict__ A,
    const _Float16* __restrict__ W0, const _Float16* __restrict__ W1,
    const float* __restrict__ bias0, const float* __restrict__ bias1,
    float* __restrict__ out0, float* __restrict__ out1)
{
    constexpr int KP = K + 8;                 // padded LDS stride (halves)
    __shared__ _Float16 sA[64 * KP];

    const int dir = blockIdx.z;
    const _Float16* W    = dir ? W1 : W0;
    const float*    bias = dir ? bias1 : bias0;
    float*          out  = dir ? out1 : out0;

    const int tid  = threadIdx.x;
    const int wave = tid >> 5;
    const int lane = tid & 31;
    const int t    = blockIdx.x;              // one WG == one time step

    // ---- async stage of the A tile (contiguous 64*K halves in global) ----
    constexpr int CH = (64 * K) / 8;          // number of 16B chunks
    const _Float16* gA = A + (size_t)t * 64 * K;
    for (int ci = tid; ci < CH; ci += 1024) {
        int row = ci / (K / 8);
        int kc8 = ci % (K / 8);
        unsigned ldsAddr = (unsigned)(uintptr_t)&sA[row * KP + kc8 * 8];
        const _Float16* src = gA + (size_t)ci * 8;
        asm volatile("global_load_async_to_lds_b128 %0, %1, off"
                     :: "v"(ldsAddr), "v"(src) : "memory");
    }
    asm volatile("s_wait_asynccnt 0" ::: "memory");
    __syncthreads();

    // ---- per-wave A fragments (batch tile m0, all K chunks) ----
    const int m0 = (wave >> 3) << 4;          // 0,16,32,48
    constexpr int NK = K / 32;
    v16h afr[NK];
    {
        const int row = m0 + (lane & 15);
        const int kb  = (lane & 16) ? 8 : 0;
        const _Float16* p = &sA[row * KP + kb];
#pragma unroll
        for (int kc = 0; kc < NK; ++kc)
            afr[kc] = make_frag(p + kc * 32, p + kc * 32 + 16);
    }

    // ---- 4 column tiles per wave (unroll 1: keep VGPR pressure low) ----
    const int jbase = (wave & 7) * 64;
    const int b0    = m0 + ((lane & 16) ? 8 : 0);
#pragma unroll 1
    for (int q = 0; q < 4; ++q) {
        const int j0 = jbase + q * 16;
        v8f acc = {};
#pragma unroll
        for (int kc = 0; kc < NK; ++kc) {
            const int kb = kc * 32 + ((lane & 16) ? 8 : 0);
            const _Float16* wp = W + (size_t)(j0 + (lane & 15)) * K + kb;
            v16h bfr = make_frag(wp, wp + 16);
            acc = __builtin_amdgcn_wmma_f32_16x16x32_f16(
                false, afr[kc], false, bfr, (short)0, acc, false, false);
        }
        const int   col = j0 + (lane & 15);
        const float bv  = bias[col];
        v8f res;
#pragma unroll
        for (int r = 0; r < 8; ++r) res[r] = acc[r] + bv;
        *(v8f*)(out + ((size_t)t * G4H + col) * BATCH + b0) = res;
    }
}

// ---------------------------------------------------------------------------
// Persistent LSTM recurrence. grid.x = 2 (dir 0=fwd, 1=bwd), 1024 threads.
// Per step: g = xg[tt] + h @ W_hh^T via WMMA; gates -> c,h in registers.
// W_hh lives entirely in LDS (139KB, converted f32->f16 at start); h is
// double-buffered in LDS. Both use stride-136 rows so all ds_load_b128
// fragment reads are bank-conflict-free. One split barrier per step:
// s_barrier_wait at top, s_wait_dscnt + s_barrier_signal after h-stores,
// letting ys stores / prefetch overlap other waves' waits. Activations use
// hardware V_TANH_F32 (5 TRANS ops per element on the critical path).
// Wave w: batch rows m0..m0+15, hidden cols u0..u0+15 across all 4 gates.
// ---------------------------------------------------------------------------
template <bool WRITE_YS, bool WRITE_HLAST>
__global__ __launch_bounds__(1024) void lstm_layer_kernel(
    const float* __restrict__ whh0, const float* __restrict__ whh1,
    const float* __restrict__ xg0, const float* __restrict__ xg1,
    _Float16* __restrict__ ys,      // [t*64+b][256] f16 (layer0)
    float* __restrict__ hlast)      // [2][64][128] f32 (layer1)
{
    constexpr int HS = HID + 8;             // padded row stride in halves
    __shared__ _Float16 sW[G4H * HS];       // 139,264 B: W_hh f16
    __shared__ _Float16 sH[2][BATCH * HS];  //  34,816 B: h double buffer

    const int dir = blockIdx.x;
    const float* WhhF = dir ? whh1 : whh0;
    const float* xg   = dir ? xg1  : xg0;

    const int tid  = threadIdx.x;
    const int wave = tid >> 5;
    const int lane = tid & 31;

    // preload W_hh (f32 global -> f16 LDS, padded rows) and zero h
    for (int i = tid; i < G4H * HID; i += 1024) {
        const int r = i >> 7, c = i & 127;
        sW[r * HS + c] = (_Float16)WhhF[i];
    }
    for (int i = tid; i < BATCH * HS; i += 1024) {
        sH[0][i] = (_Float16)0.f;
        sH[1][i] = (_Float16)0.f;
    }
    __syncthreads();
    __builtin_amdgcn_s_barrier_signal(-1);   // open round for t=0 wait

    const int m0 = (wave >> 3) << 4;     // batch tile base: 0,16,32,48
    const int u0 = (wave & 7) << 4;      // hidden column base: 0..112
    const int b0 = m0 + ((lane & 16) ? 8 : 0);
    const int ncol = lane & 15;

    v8f cc;
#pragma unroll
    for (int r = 0; r < 8; ++r) cc[r] = 0.f;

    for (int t = 0; t < TSEQ; ++t) {
        const int tt = dir ? (TSEQ - 1 - t) : t;
        const _Float16* hOld = sH[t & 1];
        _Float16*       hNew = sH[(t + 1) & 1];

        __builtin_amdgcn_s_barrier_wait(-1);   // prev step's h visible

        // ---- A fragments from h in LDS ----
        v16h afr[4];
        {
            const int row = m0 + (lane & 15);
            const int kb  = (lane & 16) ? 8 : 0;
            const _Float16* p = &hOld[row * HS + kb];
#pragma unroll
            for (int kc = 0; kc < 4; ++kc)
                afr[kc] = make_frag(p + kc * 32, p + kc * 32 + 16);
        }

        // ---- gate pre-activations: xg + h @ Whh^T (4 gate tiles) ----
        v8f ga[4];
#pragma unroll
        for (int g = 0; g < 4; ++g) {
            const int j0  = g * HID + u0;
            const int col = j0 + ncol;
            v8f acc = *(const v8f*)(xg + ((size_t)tt * G4H + col) * BATCH + b0);
#pragma unroll
            for (int kc = 0; kc < 4; ++kc) {
                const int kb = kc * 32 + ((lane & 16) ? 8 : 0);
                const _Float16* wp = &sW[col * HS + kb];
                v16h bfr = make_frag(wp, wp + 16);
                acc = __builtin_amdgcn_wmma_f32_16x16x32_f16(
                    false, afr[kc], false, bfr, (short)0, acc, false, false);
            }
            ga[g] = acc;
        }

        // ---- elementwise LSTM cell (registers only, hardware tanh) ----
        float hv[8];
#pragma unroll
        for (int r = 0; r < 8; ++r) {
            float iv = sigmoid_hw(ga[0][r]);
            float fv = sigmoid_hw(ga[1][r]);
            float gv = tanh_hw(ga[2][r]);
            float ov = sigmoid_hw(ga[3][r]);
            float c  = fv * cc[r] + iv * gv;
            cc[r] = c;
            hv[r] = ov * tanh_hw(c);
        }

        // ---- write h to LDS double buffer (+ ys / hlast to global) ----
        const int u = u0 + ncol;
#pragma unroll
        for (int r = 0; r < 8; ++r) {
            hNew[(b0 + r) * HS + u] = (_Float16)hv[r];
        }
        asm volatile("s_wait_dscnt 0" ::: "memory");  // our h-stores done
        __builtin_amdgcn_s_barrier_signal(-1);        // arrive for this step

        if (WRITE_YS) {
#pragma unroll
            for (int r = 0; r < 8; ++r)
                ys[((size_t)tt * BATCH + b0 + r) * (2 * HID) + dir * HID + u] =
                    (_Float16)hv[r];
        }
        if (WRITE_HLAST && t == TSEQ - 1) {
#pragma unroll
            for (int r = 0; r < 8; ++r)
                hlast[(size_t)dir * BATCH * HID + (b0 + r) * HID + u] = hv[r];
        }

        // prefetch next step's xg columns for this wave
        if (t + 1 < TSEQ) {
            const int ntt = dir ? (TSEQ - 2 - t) : (t + 1);
            __builtin_prefetch(xg + ((size_t)ntt * G4H + u0) * BATCH, 0, 1);
        }
    }
}

// ---------------------------------------------------------------------------
// Head: feat = [h_f, h_b] @ fc_w^T + fc_b ; out = feat @ clf_w^T + clf_b
// ---------------------------------------------------------------------------
__global__ void head_kernel(const float* __restrict__ hlast,
                            const float* __restrict__ fcw, const float* __restrict__ fcb,
                            const float* __restrict__ clfw, const float* __restrict__ clfb,
                            float* __restrict__ out)
{
    __shared__ float feat[BATCH * EMB];
    const int tid = threadIdx.x;
    for (int idx = tid; idx < BATCH * EMB; idx += 256) {
        const int b = idx >> 7, e = idx & 127;
        float s = fcb[e];
        const float* hf = hlast + (size_t)b * HID;
        const float* hb = hlast + (size_t)BATCH * HID + (size_t)b * HID;
        const float* w  = fcw + (size_t)e * (2 * HID);
        for (int u = 0; u < HID; ++u) s += hf[u] * w[u] + hb[u] * w[HID + u];
        feat[idx] = s;
    }
    __syncthreads();
    for (int idx = tid; idx < BATCH * NCLS; idx += 256) {
        const int b = idx / NCLS, n = idx % NCLS;
        float s = clfb[n];
        const float* f = feat + b * EMB;
        const float* w = clfw + (size_t)n * EMB;
        for (int e = 0; e < EMB; ++e) s += f[e] * w[e];
        out[b * NCLS + n] = s;
    }
}

// ---------------------------------------------------------------------------
// Host launcher
// ---------------------------------------------------------------------------
static inline size_t alignup(size_t x) { return (x + 255) & ~(size_t)255; }

extern "C" void kernel_launch(void* const* d_in, const int* in_sizes, int n_in,
                              void* d_out, int out_size, void* d_ws, size_t ws_size,
                              hipStream_t stream)
{
    (void)in_sizes; (void)n_in; (void)out_size; (void)ws_size;
    const float* x = (const float*)d_in[0];
    // order per layer/dir block: w_ih, w_hh, b_ih, b_hh  (l0f, l0b, l1f, l1b)
    const float* wih[4] = {(const float*)d_in[1], (const float*)d_in[5],
                           (const float*)d_in[9], (const float*)d_in[13]};
    const float* whh[4] = {(const float*)d_in[2], (const float*)d_in[6],
                           (const float*)d_in[10], (const float*)d_in[14]};
    const float* bih[4] = {(const float*)d_in[3], (const float*)d_in[7],
                           (const float*)d_in[11], (const float*)d_in[15]};
    const float* bhh[4] = {(const float*)d_in[4], (const float*)d_in[8],
                           (const float*)d_in[12], (const float*)d_in[16]};
    const float* fcw  = (const float*)d_in[17];
    const float* fcb  = (const float*)d_in[18];
    const float* clfw = (const float*)d_in[19];
    const float* clfb = (const float*)d_in[20];

    // workspace carve-up
    char* ws = (char*)d_ws;
    size_t off = 0;
    _Float16* A0 = (_Float16*)(ws + off); off = alignup(off + (size_t)MROWS * 128 * 2);
    _Float16* Wih16[4]; // l0: [512][128], l1: [512][256]
    for (int i = 0; i < 4; ++i) {
        size_t kd = (i < 2) ? 128 : 256;
        Wih16[i] = (_Float16*)(ws + off); off = alignup(off + (size_t)G4H * kd * 2);
    }
    float* biasc[4];
    for (int i = 0; i < 4; ++i) {
        biasc[i] = (float*)(ws + off); off = alignup(off + (size_t)G4H * 4);
    }
    float* xgA = (float*)(ws + off); off = alignup(off + (size_t)MROWS * G4H * 4);
    float* xgB = (float*)(ws + off); off = alignup(off + (size_t)MROWS * G4H * 4);
    _Float16* ys0 = (_Float16*)(ws + off); off = alignup(off + (size_t)MROWS * 2 * HID * 2);
    float* hlast = (float*)(ws + off); off = alignup(off + (size_t)2 * BATCH * HID * 4);

    // 1) conversions
    convert_x_kernel<<<4096, 256, 0, stream>>>(x, A0);
    convert_w_kernel<128><<<256, 256, 0, stream>>>(wih[0], Wih16[0], FIN);
    convert_w_kernel<128><<<256, 256, 0, stream>>>(wih[1], Wih16[1], FIN);
    convert_w_kernel<256><<<512, 256, 0, stream>>>(wih[2], Wih16[2], 256);
    convert_w_kernel<256><<<512, 256, 0, stream>>>(wih[3], Wih16[3], 256);
    for (int i = 0; i < 4; ++i)
        bias_kernel<<<2, 256, 0, stream>>>(bih[i], bhh[i], biasc[i]);

    // 2) layer 0: input projection (both dirs), then recurrence (both dirs)
    xproj_gemm_kernel<128><<<dim3(TSEQ, 1, 2), 1024, 0, stream>>>(
        A0, Wih16[0], Wih16[1], biasc[0], biasc[1], xgA, xgB);
    lstm_layer_kernel<true, false><<<2, 1024, 0, stream>>>(
        whh[0], whh[1], xgA, xgB, ys0, nullptr);

    // 3) layer 1: projection from ys0, then recurrence -> hlast
    xproj_gemm_kernel<256><<<dim3(TSEQ, 1, 2), 1024, 0, stream>>>(
        ys0, Wih16[2], Wih16[3], biasc[2], biasc[3], xgA, xgB);
    lstm_layer_kernel<false, true><<<2, 1024, 0, stream>>>(
        whh[2], whh[3], xgA, xgB, nullptr, hlast);

    // 4) head
    head_kernel<<<1, 256, 0, stream>>>(hlast, fcw, fcb, clfw, clfb, (float*)d_out);
}
